// FfjordModel_26053271618195
// MI455X (gfx1250) — matrix operand order repaired and checked by reading, not compiled, not run on hardware
//
#include <hip/hip_runtime.h>

// FFJORD RK4 integrator for MI455X (gfx1250, wave32).
// Compute-bound (~110 GFLOP vs ~6MB HBM): the three 64-wide matvecs per
// dynamics eval (primal L2 + two JVP streams) run on v_wmma_f32_16x16x32_f16.
// One wave = 16 batch rows; W2/W3 persist as B-fragments in VGPRs;
// activations interchange via per-wave LDS tiles (same-wave LDS is in-order,
// so no explicit dscnt flushes are needed -- only compiler barriers).
// Activations use raw v_exp/v_log/v_rcp (args in [1,2] -> no fixups); biases
// are folded into the activation input so WMMA C is inline 0.

typedef _Float16 half16_t __attribute__((ext_vector_type(16)));
typedef _Float16 half8_t  __attribute__((ext_vector_type(8)));
typedef float    f32x8_t  __attribute__((ext_vector_type(8)));

#define ROW 72                       // LDS act row stride in halves (144B, 16B aligned)
// per-wave halves: bufA + bufB + bufC + wcol(128)
#define WSLOT (3 * 16 * ROW + 128)

__device__ __forceinline__ void softsig(float x, float& sp, float& sg) {
  // softplus(x) = max(x,0) + ln(1+exp(-|x|));  sigmoid via rcp of same 1+e
  float e  = __builtin_amdgcn_exp2f(-1.442695040889f * fabsf(x)); // 1 mul + 1 trans
  float o  = 1.0f + e;                                            // o in [1,2]
  float r  = __builtin_amdgcn_rcpf(o);
  float lg = __builtin_amdgcn_logf(o);                            // log2(o)
  sp = fmaf(0.69314718056f, lg, fmaxf(x, 0.0f));
  sg = (x >= 0.0f) ? r : (1.0f - r);
}

__global__ __launch_bounds__(256) void ffjord_rk4_kernel(
    const float* __restrict__ z_in, const float* __restrict__ lp_in,
    const float* __restrict__ W1, const float* __restrict__ b1,
    const float* __restrict__ W2, const float* __restrict__ b2,
    const float* __restrict__ W3, const float* __restrict__ b3,
    float* __restrict__ out, int rows)
{
  __shared__ __align__(16) _Float16 lds[8 * WSLOT];

  const int lane = threadIdx.x & 31;
  const int wv   = threadIdx.x >> 5;
  const int ln   = lane & 15;     // N index inside 16-wide tile (C/B layout)
  const int hb   = lane >> 4;     // wave half (0/1)
  const int rowBase = (blockIdx.x * 8 + wv) * 16;
  if (rowBase >= rows) return;

  _Float16* bufA = &lds[wv * WSLOT];
  _Float16* bufB = bufA + 16 * ROW;
  _Float16* bufC = bufB + 16 * ROW;
  _Float16* wcol = bufC + 16 * ROW;   // W1 z-columns (2 x 64 f16)

  // ---------------- persistent weight fragments (f32 -> f16) ----------------
  // B fragment layout (32x16 f16): lanes 0-15 hold K=0..15 (halves 0..15),
  // lanes 16-31 hold K=16..31; N = lane&15.
  half16_t BW2[2][4];              // layer2 z-part: 64x64
  #pragma unroll
  for (int kb = 0; kb < 2; ++kb)
    #pragma unroll
    for (int nb = 0; nb < 4; ++nb) {
      half16_t b;
      int n = nb * 16 + ln, k0 = kb * 32 + hb * 16;
      #pragma unroll
      for (int e = 0; e < 16; ++e) b[e] = (_Float16)W2[n * 65 + 1 + k0 + e];
      BW2[kb][nb] = b;
    }
  half16_t BW3[2];                 // layer3 z-part: 64x2 (cols 0,1 of tile)
  #pragma unroll
  for (int kb = 0; kb < 2; ++kb) {
    half16_t b;
    int k0 = kb * 32 + hb * 16;
    #pragma unroll
    for (int e = 0; e < 16; ++e)
      b[e] = (ln < 2) ? (_Float16)W3[ln * 65 + 1 + k0 + e] : (_Float16)0.0f;
    BW3[kb] = b;
  }
  // W1 z-columns into per-wave LDS (tangent seeds, loaded per eval)
  for (int k = lane; k < 64; k += 32) {
    wcol[k]      = (_Float16)W1[k * 3 + 1];
    wcol[64 + k] = (_Float16)W1[k * 3 + 2];
  }
  // per-lane scalar weights for layer 1 (VALU, K=2) and bias/time columns
  float w1a[4], w1b[4], w10[4], b1v[4], w20[4], b2v[4];
  #pragma unroll
  for (int tc = 0; tc < 4; ++tc) {
    int n = tc * 16 + ln;
    w1a[tc] = W1[n * 3 + 1]; w1b[tc] = W1[n * 3 + 2];
    w10[tc] = W1[n * 3];     b1v[tc] = b1[n];
    w20[tc] = W2[n * 65];    b2v[tc] = b2[n];
  }
  const float w30 = (ln < 2) ? W3[ln * 65] : 0.0f;
  const float b3v = (ln < 2) ? b3[ln] : 0.0f;

  // ---------------- state (C layout: lane col = ln, reg v -> row v+8*hb) ----
  float zc[8], lp[8];
  #pragma unroll
  for (int v = 0; v < 8; ++v) {
    int m = rowBase + v + 8 * hb;
    zc[v] = (ln < 2) ? z_in[m * 2 + ln] : 0.0f;
    lp[v] = (ln == 0) ? lp_in[m] : 0.0f;
  }

  auto zerov = [&]() -> f32x8_t {
    f32x8_t z;
    #pragma unroll
    for (int r = 0; r < 8; ++r) z[r] = 0.0f;
    return z;
  };
  auto loadFrag = [&](const _Float16* buf, int row, int kb) -> half16_t {
    int base = row * ROW + kb * 32 + hb * 16;
    half8_t lo = *reinterpret_cast<const half8_t*>(buf + base);
    half8_t hi = *reinterpret_cast<const half8_t*>(buf + base + 8);
    return __builtin_shufflevector(lo, hi, 0,1,2,3,4,5,6,7,8,9,10,11,12,13,14,15);
  };

  const int shBase = lane & 16;   // source lane base for same-half broadcast

  // ---------------- dynamics eval: zdot = MLP(t,z); ldot = -div --------------
  auto evalDyn = [&](float t, const float* zs, float* zdot, float* ldot) {
    // broadcast z components (C-layout cols 0,1) to all lanes of this half
    float z0[8], z1[8];
    #pragma unroll
    for (int r = 0; r < 8; ++r) {
      z0[r] = __shfl(zs[r], shBase, 32);
      z1[r] = __shfl(zs[r], shBase + 1, 32);
    }

    // layer 1 (K=2, VALU): x = W1z@z + (W1[:,0]*t + b1); softplus->A, sigmoid->B
    #pragma unroll
    for (int tc = 0; tc < 4; ++tc) {
      float tb1 = fmaf(w10[tc], t, b1v[tc]);
      #pragma unroll
      for (int r = 0; r < 8; ++r) {
        float x = fmaf(w1a[tc], z0[r], fmaf(w1b[tc], z1[r], tb1));
        float sp, sg; softsig(x, sp, sg);
        int m = r + 8 * hb, n = tc * 16 + ln;
        bufA[m * ROW + n] = (_Float16)sp;
        bufB[m * ROW + n] = (_Float16)sg;
      }
    }
    // same-wave LDS is in-order: compiler barrier only (no dscnt flush)
    asm volatile("" ::: "memory");
    half16_t Ah1[2], At10[2], At11[2];
    #pragma unroll
    for (int kb = 0; kb < 2; ++kb) {
      Ah1[kb] = loadFrag(bufA, ln, kb);
      half16_t s1 = loadFrag(bufB, ln, kb);          // sigmoid(h1_pre)
      At10[kb] = s1 * loadFrag(wcol, 0, kb);         // JVP seed, dim 0
      At11[kb] = s1 * loadFrag(wcol + 64, 0, kb);    // JVP seed, dim 1
    }
    asm volatile("" ::: "memory");

    // layer 2 (K=64): primal + both tangent streams share B fragments.
    // Per 16-col tile: 6 WMMAs, then activation; h2->A, t2_0->B, t2_1->C.
    #pragma unroll
    for (int tc = 0; tc < 4; ++tc) {
      float tb2 = fmaf(w20[tc], t, b2v[tc]);
      f32x8_t cP = zerov(), c0 = zerov(), c1 = zerov();
      cP = __builtin_amdgcn_wmma_f32_16x16x32_f16(false, Ah1[0],  false, BW2[0][tc], (short)0, cP, false, false);
      cP = __builtin_amdgcn_wmma_f32_16x16x32_f16(false, Ah1[1],  false, BW2[1][tc], (short)0, cP, false, false);
      c0 = __builtin_amdgcn_wmma_f32_16x16x32_f16(false, At10[0], false, BW2[0][tc], (short)0, c0, false, false);
      c0 = __builtin_amdgcn_wmma_f32_16x16x32_f16(false, At10[1], false, BW2[1][tc], (short)0, c0, false, false);
      c1 = __builtin_amdgcn_wmma_f32_16x16x32_f16(false, At11[0], false, BW2[0][tc], (short)0, c1, false, false);
      c1 = __builtin_amdgcn_wmma_f32_16x16x32_f16(false, At11[1], false, BW2[1][tc], (short)0, c1, false, false);
      #pragma unroll
      for (int r = 0; r < 8; ++r) {
        float x = cP[r] + tb2;
        float sp, sg; softsig(x, sp, sg);
        int m = r + 8 * hb, n = tc * 16 + ln;
        bufA[m * ROW + n] = (_Float16)sp;            // h2
        bufB[m * ROW + n] = (_Float16)(c0[r] * sg);  // t2_0 = sig2 * t2pre_0
        bufC[m * ROW + n] = (_Float16)(c1[r] * sg);  // t2_1
      }
    }
    asm volatile("" ::: "memory");
    half16_t Ah2[2], At20[2], At21[2];
    #pragma unroll
    for (int kb = 0; kb < 2; ++kb) {
      Ah2[kb]  = loadFrag(bufA, ln, kb);
      At20[kb] = loadFrag(bufB, ln, kb);
      At21[kb] = loadFrag(bufC, ln, kb);
    }
    asm volatile("" ::: "memory");

    // layer 3 (64 -> 2): zdot in cols 0,1; divergence from tangent cols
    f32x8_t zd = zerov(), d0 = zerov(), d1 = zerov();
    zd = __builtin_amdgcn_wmma_f32_16x16x32_f16(false, Ah2[0],  false, BW3[0], (short)0, zd, false, false);
    zd = __builtin_amdgcn_wmma_f32_16x16x32_f16(false, Ah2[1],  false, BW3[1], (short)0, zd, false, false);
    d0 = __builtin_amdgcn_wmma_f32_16x16x32_f16(false, At20[0], false, BW3[0], (short)0, d0, false, false);
    d0 = __builtin_amdgcn_wmma_f32_16x16x32_f16(false, At20[1], false, BW3[1], (short)0, d0, false, false);
    d1 = __builtin_amdgcn_wmma_f32_16x16x32_f16(false, At21[0], false, BW3[0], (short)0, d1, false, false);
    d1 = __builtin_amdgcn_wmma_f32_16x16x32_f16(false, At21[1], false, BW3[1], (short)0, d1, false, false);

    float tb3 = fmaf(w30, t, b3v);
    #pragma unroll
    for (int r = 0; r < 8; ++r) {
      zdot[r] = zd[r] + tb3;                      // valid lanes ln<2
      float dd1 = __shfl_xor(d1[r], 1, 32);       // bring col-1 value to col 0
      ldot[r] = -(d0[r] + dd1);                   // valid lanes ln==0
    }
  };

  // ---------------- RK4, 4 fixed steps over t in [0,1] ----------------------
  const float dtt = 0.25f;
  #pragma clang loop unroll(disable)
  for (int step = 0; step < 4; ++step) {
    float t0 = (float)step * dtt;
    float zs[8], kz[8], kl[8], kzs[8], kls[8];
    #pragma unroll
    for (int r = 0; r < 8; ++r) { zs[r] = zc[r]; kzs[r] = 0.0f; kls[r] = 0.0f; }
    #pragma clang loop unroll(disable)
    for (int s = 0; s < 4; ++s) {
      float tf = (s == 0) ? 0.0f : ((s == 3) ? 1.0f : 0.5f);
      float wt = (s == 1 || s == 2) ? 2.0f : 1.0f;
      evalDyn(t0 + tf * dtt, zs, kz, kl);
      float cn = (s < 2) ? 0.5f * dtt : dtt;
      #pragma unroll
      for (int r = 0; r < 8; ++r) {
        kzs[r] = fmaf(wt, kz[r], kzs[r]);
        kls[r] = fmaf(wt, kl[r], kls[r]);
        zs[r]  = fmaf(cn, kz[r], zc[r]);
      }
    }
    #pragma unroll
    for (int r = 0; r < 8; ++r) {
      zc[r] = fmaf(dtt / 6.0f, kzs[r], zc[r]);
      lp[r] = fmaf(dtt / 6.0f, kls[r], lp[r]);
    }
  }

  // ---------------- store zf (B x 2) then lf (B x 1), concatenated ----------
  #pragma unroll
  for (int v = 0; v < 8; ++v) {
    int m = rowBase + v + 8 * hb;
    if (ln < 2)  out[m * 2 + ln] = zc[v];
    if (ln == 0) out[2 * rows + m] = lp[v];
  }
}

extern "C" void kernel_launch(void* const* d_in, const int* in_sizes, int n_in,
                              void* d_out, int out_size, void* d_ws, size_t ws_size,
                              hipStream_t stream) {
  const float* z  = (const float*)d_in[0];
  const float* lp = (const float*)d_in[1];
  const float* W1 = (const float*)d_in[2];
  const float* b1 = (const float*)d_in[3];
  const float* W2 = (const float*)d_in[4];
  const float* b2 = (const float*)d_in[5];
  const float* W3 = (const float*)d_in[6];
  const float* b3 = (const float*)d_in[7];
  float* out = (float*)d_out;
  const int rows   = in_sizes[0] / 2;        // 262144
  const int blocks = (rows + 127) / 128;     // 8 waves x 16 rows per block
  ffjord_rk4_kernel<<<blocks, 256, 0, stream>>>(z, lp, W1, b1, W2, b2, W3, b3, out, rows);
}
